// Self_Attention_56547539419343
// MI455X (gfx1250) — matrix operand order repaired and checked by reading, not compiled, not run on hardware
//
#include <hip/hip_runtime.h>
#include <hip/hip_bf16.h>

// CDNA5 / gfx1250 flash-style self-attention, f16 WMMA with fp32 accumulation.
// Pipeline (all on `stream`):
//   0a) transpose-convert x[N,C,HW] f32 -> xhT[N,HW,C] f16 (LDS tiled)
//   0b) convert Wq/Wk/Wv/Wo f32 -> f16
//   1)  proj GEMMs: qT/kT[N,HW,32] f16, vh[N,256,HW] f16   (WMMA 16x16x32 f16)
//   2)  row stats: online max / sum of exp over j  (one WMMA per 16x16 e-tile)
//   3)  attention: recompute e tiles, p=exp(e-m)/s staged in LDS, o = v @ p
//       (8 waves/WG: 1 e-WMMA + 8 o-WMMAs per wave per 32-i chunk)
//   4)  out = gamma * Wo @ o  (WMMA, fp32 store)
// Needs ~53 MB workspace.

typedef __attribute__((ext_vector_type(16))) _Float16 v16h;
typedef __attribute__((ext_vector_type(8)))  _Float16 v8h;
typedef __attribute__((ext_vector_type(8)))  float    v8f;

#define L2E 1.44269504088896340736f

static constexpr int NB = 8, C = 256, HW = 4096, CQ = 32, CV = 256;

// ---- WMMA fragment helpers (wave32; A/B k order contiguous (k = 16*hi + e), D layout per CDNA5 ISA 7.12.2) ----

// A 16x32 f16, M=lane&15, K contiguous in memory at `row`:
// elems e = K[hi*16 + e]
__device__ __forceinline__ v16h loadA16x32(const _Float16* row, int hi) {
  return *(const v16h*)(row + hi * 16);
}
// B 32x16 f16, N=lane&15, elems e = K[hi*16 + e] (K contiguous at `col`)
__device__ __forceinline__ v16h loadB32x16(const _Float16* col, int hi) {
  return *(const v16h*)(col + hi * 16);
}
__device__ __forceinline__ v8f wmma_f16(v16h a, v16h b, v8f c) {
  return __builtin_amdgcn_wmma_f32_16x16x32_f16(false, a, false, b, (short)0, c,
                                                false, false);
}
// D f32 elem r <-> row (8*hi + r), col (lane&15); pack rows to 8 consecutive f16
__device__ __forceinline__ void store_half8(_Float16* dst, v8f d) {
  v8h h = {};
#pragma unroll
  for (int r = 0; r < 8; ++r) h[r] = (_Float16)d[r];
  *(v8h*)dst = h;
}

// ---- 0a: x[N,C,HW] f32 -> xhT[N,HW,C] f16 (LDS tile transpose) ----
__global__ __launch_bounds__(256) void xpose_kernel(const float* __restrict__ x,
                                                    _Float16* __restrict__ xhT) {
  __shared__ float t[32][33];
  const int n = blockIdx.z, c0 = blockIdx.y * 32, l0 = blockIdx.x * 32;
  const int tx = threadIdx.x, ty = threadIdx.y;
  const float* xb = x + (size_t)n * C * HW;
#pragma unroll
  for (int k = 0; k < 32; k += 8)
    t[ty + k][tx] = xb[(size_t)(c0 + ty + k) * HW + l0 + tx];
  __syncthreads();
  _Float16* ob = xhT + (size_t)n * HW * C;
#pragma unroll
  for (int k = 0; k < 32; k += 8)
    ob[(size_t)(l0 + ty + k) * C + c0 + tx] = (_Float16)t[tx][ty + k];
}

// ---- 0b: f32 -> f16 ----
__global__ __launch_bounds__(256) void cvt_f16_kernel(const float* __restrict__ s,
                                                      _Float16* __restrict__ d, int n) {
  int i = blockIdx.x * 256 + threadIdx.x;
  if (i < n) d[i] = (_Float16)s[i];
}

// ---- 1: projection GEMM. out = Wh[M,C] @ xhT[n,:,:]^T, M = MT*16 ----
// modeT=1: store transposed outT[n][HW][M] (for qT/kT); modeT=0: out[n][M][HW] (v)
__global__ __launch_bounds__(256) void proj_kernel(const _Float16* __restrict__ xhT,
                                                   const _Float16* __restrict__ Wh,
                                                   _Float16* __restrict__ out,
                                                   int MT, int modeT) {
  const int lane = threadIdx.x & 31, m = lane & 15, hi = lane >> 4;
  int tile = blockIdx.x * 8 + (threadIdx.x >> 5);
  const int it = tile & 255; tile >>= 8;
  const int mt = tile % MT;
  const int n  = tile / MT;
  const int i0 = it * 16;
  v8f acc = {0.f, 0.f, 0.f, 0.f, 0.f, 0.f, 0.f, 0.f};
#pragma unroll
  for (int k0 = 0; k0 < C; k0 += 32) {
    v16h aW = loadA16x32(Wh + (size_t)(mt * 16 + m) * C + k0, hi);
    v16h bX = loadB32x16(xhT + ((size_t)n * HW + i0 + m) * C + k0, hi);
    acc = wmma_f16(aW, bX, acc);
  }
  const int Mr = MT * 16;
  if (modeT) {
    store_half8(out + ((size_t)n * HW + i0 + m) * Mr + mt * 16 + 8 * hi, acc);
  } else {
#pragma unroll
    for (int r = 0; r < 8; ++r)
      out[((size_t)n * Mr + mt * 16 + 8 * hi + r) * HW + i0 + m] = (_Float16)acc[r];
  }
}

// ---- 2: per-row softmax stats over all j: rowm = max_j e, srcp = 1/sum exp ----
__global__ __launch_bounds__(256) void rowstats_kernel(const _Float16* __restrict__ qT,
                                                       const _Float16* __restrict__ kT,
                                                       float* __restrict__ rowm,
                                                       float* __restrict__ srcp) {
  const int lane = threadIdx.x & 31, m = lane & 15, hi = lane >> 4;
  const int wave = blockIdx.x * 8 + (threadIdx.x >> 5);
  const int n = wave >> 8, iblk = wave & 255;
  const v8f zero = {0.f, 0.f, 0.f, 0.f, 0.f, 0.f, 0.f, 0.f};
  v16h aq = loadA16x32(qT + ((size_t)n * HW + iblk * 16 + m) * CQ, hi);
  float rm[8], rs[8];
#pragma unroll
  for (int r = 0; r < 8; ++r) { rm[r] = -__builtin_inff(); rs[r] = 0.f; }
  for (int jt = 0; jt < 256; ++jt) {
    v16h bk = loadB32x16(kT + ((size_t)n * HW + jt * 16 + m) * CQ, hi);
    v8f e = wmma_f16(aq, bk, zero);
#pragma unroll
    for (int r = 0; r < 8; ++r) {
      float ev = e[r];
      float nm = fmaxf(rm[r], ev);
      rs[r] = rs[r] * exp2f((rm[r] - nm) * L2E) + exp2f((ev - nm) * L2E);
      rm[r] = nm;
    }
  }
  // combine partials across the 16 lanes holding the same rows
#pragma unroll
  for (int mask = 1; mask < 16; mask <<= 1) {
#pragma unroll
    for (int r = 0; r < 8; ++r) {
      float om = __shfl_xor(rm[r], mask, 32);
      float os = __shfl_xor(rs[r], mask, 32);
      float nm = fmaxf(rm[r], om);
      rs[r] = rs[r] * exp2f((rm[r] - nm) * L2E) + os * exp2f((om - nm) * L2E);
      rm[r] = nm;
    }
  }
  if (m == 0) {
#pragma unroll
    for (int r = 0; r < 8; ++r) {
      int i = iblk * 16 + 8 * hi + r;
      rowm[(size_t)n * HW + i] = rm[r];
      srcp[(size_t)n * HW + i] = 1.0f / rs[r];
    }
  }
}

// ---- 3: o[vc,j] = sum_i v[vc,i] * exp(e[i,j]-m[i])/s[i];  store oT[n][j][vc] f16
__global__ __launch_bounds__(256) void attn_kernel(const _Float16* __restrict__ qT,
                                                   const _Float16* __restrict__ kT,
                                                   const _Float16* __restrict__ vh,
                                                   const float* __restrict__ rowm,
                                                   const float* __restrict__ srcp,
                                                   _Float16* __restrict__ oT) {
  __shared__ _Float16 pT[64 * 32];  // p transposed: [j_local 64][i_local 32]
  const int n = blockIdx.x >> 6;
  const int j0 = (blockIdx.x & 63) * 64;
  const int wid = threadIdx.x >> 5;
  const int lane = threadIdx.x & 31, m = lane & 15, hi = lane >> 4;
  const int is = wid >> 2, js = wid & 3;   // this wave's e sub-tile
  const int vcb = wid * 32;                // this wave's 32 v-channel rows
  const v8f zero = {0.f, 0.f, 0.f, 0.f, 0.f, 0.f, 0.f, 0.f};

  // k B-fragment fixed for the wave's j columns across all i chunks
  v16h bk = loadB32x16(kT + ((size_t)n * HW + j0 + js * 16 + m) * CQ, hi);

  v8f acc[2][4];
#pragma unroll
  for (int mt = 0; mt < 2; ++mt)
#pragma unroll
    for (int nt = 0; nt < 4; ++nt) acc[mt][nt] = zero;

  for (int ib = 0; ib < HW; ib += 32) {
    // e tile (16 i x 16 j), K = Cq = 32 in a single WMMA
    v16h aq = loadA16x32(qT + ((size_t)n * HW + ib + is * 16 + m) * CQ, hi);
    v8f e = wmma_f16(aq, bk, zero);
    v8h pv = {};
#pragma unroll
    for (int r = 0; r < 8; ++r) {
      int irow = ib + is * 16 + 8 * hi + r;
      float mm = rowm[(size_t)n * HW + irow];
      float sc = srcp[(size_t)n * HW + irow];
      pv[r] = (_Float16)(exp2f((e[r] - mm) * L2E) * sc);
    }
    *(v8h*)&pT[(js * 16 + m) * 32 + is * 16 + 8 * hi] = pv;
    __syncthreads();
#pragma unroll
    for (int mt = 0; mt < 2; ++mt) {
      v16h av = loadA16x32(vh + ((size_t)n * CV + vcb + mt * 16 + m) * HW + ib, hi);
#pragma unroll
      for (int nt = 0; nt < 4; ++nt) {
        v16h pb = *(const v16h*)&pT[(nt * 16 + m) * 32 + hi * 16];
        acc[mt][nt] = wmma_f16(av, pb, acc[mt][nt]);
      }
    }
    __syncthreads();
  }
#pragma unroll
  for (int mt = 0; mt < 2; ++mt)
#pragma unroll
    for (int nt = 0; nt < 4; ++nt)
      store_half8(oT + ((size_t)n * HW + j0 + nt * 16 + m) * CV + vcb + mt * 16 + 8 * hi,
                  acc[mt][nt]);
}

// ---- 4: out[n,oc,l] = gamma * sum_vc Wo[oc,vc] * o[n,vc,l]  (fp32 out) ----
__global__ __launch_bounds__(256) void out_kernel(const _Float16* __restrict__ Woh,
                                                  const _Float16* __restrict__ oT,
                                                  const float* __restrict__ gamma,
                                                  float* __restrict__ out) {
  const int lane = threadIdx.x & 31, m = lane & 15, hi = lane >> 4;
  int tile = blockIdx.x * 8 + (threadIdx.x >> 5);
  const int jt = tile & 255; tile >>= 8;
  const int mt = tile & 15;
  const int n = tile >> 4;
  v8f acc = {0.f, 0.f, 0.f, 0.f, 0.f, 0.f, 0.f, 0.f};
#pragma unroll
  for (int k0 = 0; k0 < CV; k0 += 32) {
    v16h aW = loadA16x32(Woh + (size_t)(mt * 16 + m) * CV + k0, hi);
    v16h bO = loadB32x16(oT + ((size_t)n * HW + jt * 16 + m) * CV + k0, hi);
    acc = wmma_f16(aW, bO, acc);
  }
  float g = gamma[0];
#pragma unroll
  for (int r = 0; r < 8; ++r)
    out[((size_t)n * C + mt * 16 + 8 * hi + r) * HW + jt * 16 + m] = g * acc[r];
}

extern "C" void kernel_launch(void* const* d_in, const int* in_sizes, int n_in,
                              void* d_out, int out_size, void* d_ws, size_t ws_size,
                              hipStream_t stream) {
  const float* x  = (const float*)d_in[0];
  const float* Wq = (const float*)d_in[1];
  const float* Wk = (const float*)d_in[2];
  const float* Wv = (const float*)d_in[3];
  const float* Wo = (const float*)d_in[4];
  const float* gm = (const float*)d_in[5];
  float* out = (float*)d_out;
  (void)in_sizes; (void)n_in; (void)out_size; (void)ws_size;

  char* ws = (char*)d_ws;
  size_t off = 0;
  auto take = [&](size_t bytes) -> char* {
    char* p = ws + off;
    off = (off + bytes + 255) & ~(size_t)255;
    return p;
  };
  _Float16* xhT = (_Float16*)take((size_t)NB * HW * C * 2);   // 16 MB
  _Float16* Wqh = (_Float16*)take((size_t)CQ * C * 2);
  _Float16* Wkh = (_Float16*)take((size_t)CQ * C * 2);
  _Float16* Wvh = (_Float16*)take((size_t)CV * C * 2);
  _Float16* Woh = (_Float16*)take((size_t)C * CV * 2);
  _Float16* qT  = (_Float16*)take((size_t)NB * HW * CQ * 2);  // 2 MB
  _Float16* kT  = (_Float16*)take((size_t)NB * HW * CQ * 2);  // 2 MB
  _Float16* vh  = (_Float16*)take((size_t)NB * CV * HW * 2);  // 16 MB
  float*    rowm = (float*)take((size_t)NB * HW * 4);
  float*    srcp = (float*)take((size_t)NB * HW * 4);
  _Float16* oT  = (_Float16*)take((size_t)NB * HW * CV * 2);  // 16 MB

  xpose_kernel<<<dim3(HW / 32, C / 32, NB), dim3(32, 8), 0, stream>>>(x, xhT);
  cvt_f16_kernel<<<(CQ * C + 255) / 256, 256, 0, stream>>>(Wq, Wqh, CQ * C);
  cvt_f16_kernel<<<(CQ * C + 255) / 256, 256, 0, stream>>>(Wk, Wkh, CQ * C);
  cvt_f16_kernel<<<(CV * C + 255) / 256, 256, 0, stream>>>(Wv, Wvh, CV * C);
  cvt_f16_kernel<<<(C * CV + 255) / 256, 256, 0, stream>>>(Wo, Woh, C * CV);

  proj_kernel<<<NB * 2 * 256 / 8, 256, 0, stream>>>(xhT, Wqh, qT, 2, 1);
  proj_kernel<<<NB * 2 * 256 / 8, 256, 0, stream>>>(xhT, Wkh, kT, 2, 1);
  proj_kernel<<<NB * 16 * 256 / 8, 256, 0, stream>>>(xhT, Wvh, vh, 16, 0);

  rowstats_kernel<<<NB * 256 / 8, 256, 0, stream>>>(qT, kT, rowm, srcp);
  attn_kernel<<<NB * 64, 256, 0, stream>>>(qT, kT, vh, rowm, srcp, oT);
  out_kernel<<<NB * 16 * 256 / 8, 256, 0, stream>>>(Woh, oT, gm, out);
}
